// NCutLossOptimized_3083786518689
// MI455X (gfx1250) — matrix operand order repaired and checked by reading, not compiled, not run on hardware
//
#include <hip/hip_runtime.h>
#include <hip/hip_bf16.h>
#include <math.h>

// NCut loss as banded GEMM on CDNA5 WMMA (gfx1250, wave32).
//   Per 16-pixel row tile: D(16x16) += A(16x768,f16) x B(768x16,f16)
//   A = affinity weights over the 21x36 support band (branch-free selects),
//   B cols 0..3 = label planes at support pixels, col 4 = ones -> den_map.
//   f32 accumulation. Support cursors tracked incrementally (no int div).
//   Epilogue: D spilled to padded LDS tile, uniform 8-lane reduction.

typedef __attribute__((ext_vector_type(16))) _Float16 v16h;
typedef __attribute__((ext_vector_type(8)))  float    v8f;

#define RADIUS   10
#define HH       256
#define WW       256
#define KC       4
#define SUP_ROWS 21
#define SUP_COLS 36
#define SUP_N    (SUP_ROWS * SUP_COLS)   // 756 valid support pixels
#define SUP_PAD  768                     // padded to 24 chunks of K=32
#define NCHUNK   24
// Gaussian constants with log2(e) folded in: exp(x) = exp2(x*log2e)
#define C_SX     (-0.09016844005555896f)   // -log2(e)/sigma_x^2 (16)
#define C_SI     (-0.0016029944898766259f) // -log2(e)/sigma_i^2 (900)
#define WAVES_PER_BLOCK 4

__global__ void ncut_zero_ws(float* ws8) {
    if (threadIdx.x < 8) ws8[threadIdx.x] = 0.0f;
}

// advance a support cursor by d (< SUP_COLS): at most one column wrap
__device__ __forceinline__ void adv(int& s, int& sr, int& sc, int d) {
    s  += d;
    sc += d;
    if (sc >= SUP_COLS) { sc -= SUP_COLS; sr += 1; }
}

__global__ __launch_bounds__(WAVES_PER_BLOCK * 32)
void ncut_tile_wmma(const float* __restrict__ img,
                    const float* __restrict__ lab,
                    float* __restrict__ ws8) {
    __shared__ float sI[WAVES_PER_BLOCK][SUP_PAD];   // support intensities/wave
    __shared__ float Dt[WAVES_PER_BLOCK][16][17];    // D spill, +1 pad vs banks
    __shared__ float acc[8];                          // num[0..3], den[0..3]

    const int tid  = threadIdx.x;
    const int wave = tid >> 5;
    const int lane = tid & 31;

    const int tile = blockIdx.x * WAVES_PER_BLOCK + wave;  // 0..4095
    const int ty   = tile >> 4;          // image row of this tile
    const int tx0  = (tile & 15) << 4;   // first column of the 16-pixel tile

    if (tid < 8) acc[tid] = 0.0f;

    __builtin_prefetch(&lab[ty * WW + tx0], 0, 3);  // global_prefetch_b8

    // ---- Stage support intensities (21x36 band, zero-padded) into LDS ----
    float* my = sI[wave];
    for (int it = 0; it < NCHUNK; ++it) {
        int idx = it * 32 + lane;        // 0..767
        float v = 0.0f;
        if (idx < SUP_N) {
            int sr = idx / SUP_COLS;
            int sc = idx - sr * SUP_COLS;
            int sy = ty - RADIUS + sr;
            int sx = tx0 - RADIUS + sc;
            if (sy >= 0 && sy < HH && sx >= 0 && sx < WW) v = img[sy * WW + sx];
        }
        my[idx] = v;                     // padded tail written as 0
    }
    __syncthreads();

    const int   m  = lane & 15;          // A row (tile pixel) this lane owns
    const int   hi = lane >> 4;          // lane-half selector for K mapping
    const int   n  = m;                  // B/D column this lane owns
    const float Ic = img[ty * WW + tx0 + m];

    // Lane-constant B selectors (branch-free B fill)
    const float selA = (n < KC)  ? 1.0f : 0.0f;   // gated label load
    const float selB = (n == KC) ? 1.0f : 0.0f;   // ones column (den_map)
    const float* __restrict__ labn = lab + ((n < KC) ? n : 0) * (HH * WW);

    // Incremental support cursors (no integer division in the hot loop).
    int sA = 8 * hi,  srA = 0, scA = 8 * hi;
    int sB = 8 * hi,  srB = 0, scB = 8 * hi;

    v8f c = {};                          // f32 accumulator fragment
    for (int ch = 0; ch < NCHUNK; ++ch) {
        v16h a, b;

        // ---- A fragment (16x32 f16), ISA layout:
        //   halves 0..7 -> K = 8*hi + i ; halves 8..15 -> K = 16 + 8*hi + i-8
        //   slot deltas: 1 x7, 9, 1 x7, (9 to next chunk)
        #pragma unroll
        for (int i = 0; i < 16; ++i) {
            int dy  = srA - RADIUS;
            int dxm = scA - RADIUS - m;
            int sy  = ty + dy;
            int sx  = tx0 - RADIUS + scA;
            bool ok = (srA < SUP_ROWS) &
                      ((unsigned)(scA - m) < (unsigned)SUP_ROWS) &  // |dx|<=10
                      ((unsigned)sy < (unsigned)HH) &
                      ((unsigned)sx < (unsigned)WW);
            float di = my[sA] - Ic;                     // unconditional ds_load
            int   ii = dy * dy + dxm * dxm;
            float e2 = fmaf(di * di, C_SI, (float)ii * C_SX);
            float w  = __builtin_amdgcn_exp2f(e2);      // always finite
            w = ok ? w : 0.0f;                          // single v_cndmask
            a[i] = (_Float16)w;
            adv(sA, srA, scA, (i == 7) ? 9 : ((i == 15) ? 9 : 1));
        }

        // ---- B fragment (32x16 f16): column n = lane&15, K = 8*hi + i (i<8), 16 + 8*hi + i-8 (i>=8).
        //   slot deltas: 1 x7, 9, 1 x7, (9 to next chunk). Branch-free masked load.
        #pragma unroll
        for (int i = 0; i < 16; ++i) {
            int sy = ty - RADIUS + srB;
            int sx = tx0 - RADIUS + scB;
            bool inb = (srB < SUP_ROWS) &
                       ((unsigned)sy < (unsigned)HH) &
                       ((unsigned)sx < (unsigned)WW);
            int   off = inb ? (sy * WW + sx) : 0;       // clamped address
            float ld  = labn[off];                      // unconditional load
            float g   = inb ? selA : 0.0f;
            b[i] = (_Float16)fmaf(g, ld, selB);
            adv(sB, srB, scB, (i == 7) ? 9 : ((i == 15) ? 9 : 1));
        }

        // (neg_a, A, neg_b, B, c_mod, C, reuse_a, reuse_b)
        c = __builtin_amdgcn_wmma_f32_16x16x32_f16(
                false, a, false, b, (short)0, c, false, false);
    }

    // ---- Spill D to LDS (uniform, conflict-free via +17 row stride) ----
    // D layout: VGPR r holds M = r + 8*hi, column N = lane&15.
    #pragma unroll
    for (int r = 0; r < 8; ++r) {
        Dt[wave][r + 8 * hi][n] = c[r];
    }
    __syncthreads();

    // ---- Uniform epilogue: lanes 0..7 own one accumulator each.
    //   lane j<4:  num_j = sum_m p_j[q]*num_map_j[q]   (Dt col j)
    //   lane j>=4: den_{j-4} = sum_m p_{j-4}[q]*den_map[q] (Dt col 4)
    if (lane < 8) {
        const int   k    = lane & 3;
        const int   col  = (lane < 4) ? k : KC;
        const float* pk  = lab + k * (HH * WW) + ty * WW + tx0;
        float s = 0.0f;
        #pragma unroll
        for (int mm = 0; mm < 16; ++mm) {
            s = fmaf(pk[mm], Dt[wave][mm][col], s);
        }
        atomicAdd(&acc[lane], s);
    }
    __syncthreads();
    if (tid < 8) atomicAdd(&ws8[tid], acc[tid]);   // one global atomic / block
}

__global__ void ncut_finalize(const float* __restrict__ ws8,
                              float* __restrict__ out) {
    if (threadIdx.x == 0) {
        float s = 0.0f;
        #pragma unroll
        for (int k = 0; k < KC; ++k) s += fabsf(ws8[k] / ws8[4 + k]);
        out[0] = (float)KC - s;
    }
}

extern "C" void kernel_launch(void* const* d_in, const int* in_sizes, int n_in,
                              void* d_out, int out_size, void* d_ws, size_t ws_size,
                              hipStream_t stream) {
    const float* img = (const float*)d_in[0];   // [1,1,256,256] f32
    const float* lab = (const float*)d_in[1];   // [1,4,256,256] f32
    float* ws8 = (float*)d_ws;                  // 8 floats: num[4], den[4]
    float* out = (float*)d_out;                 // scalar f32

    ncut_zero_ws<<<1, 32, 0, stream>>>(ws8);
    // 4096 tiles (256 rows x 16 tiles/row), 1 wave32/tile, 4 waves/block.
    ncut_tile_wmma<<<4096 / WAVES_PER_BLOCK, WAVES_PER_BLOCK * 32, 0, stream>>>(
        img, lab, ws8);
    ncut_finalize<<<1, 1, 0, stream>>>(ws8, out);
}
